// convlstm_cell_35373350649913
// MI455X (gfx1250) — compile-verified, matrix operand order checked by reading, not run on hardware
//
#include <hip/hip_runtime.h>
#include <hip/hip_bf16.h>

// ---------------------------------------------------------------------------
// ConvLSTM cell with dual self-attention for MI455X (gfx1250).
// Attention matmuls use v_wmma_f32_16x16x32_f16 (wave32 WMMA), flash-attention
// streaming softmax so the 4096x4096 attention matrix never hits memory.
// Softmax row reductions use v_permlane16_b32 (VALU) instead of ds_bpermute
// so the DS pipe is reserved for the P-tile layout round-trip.
// ---------------------------------------------------------------------------

typedef __attribute__((ext_vector_type(16))) _Float16 v16h;
typedef __attribute__((ext_vector_type(8)))  _Float16 v8h;
typedef __attribute__((ext_vector_type(8)))  float    v8f;

// Lane swizzle within each 16-lane row (VALU v_permlane16_b32, no DS pipe).
__device__ __forceinline__ float permrow16(float v, unsigned lo, unsigned hi) {
    const int i = __builtin_bit_cast(int, v);
    const int p = __builtin_amdgcn_permlane16(i, i, lo, hi, false, false);
    return __builtin_bit_cast(float, p);
}
// Butterfly (xor 1,3,7,8) max/sum across a 16-lane row.
__device__ __forceinline__ float rowred_max(float v) {
    v = fmaxf(v, permrow16(v, 0x67452301u, 0xEFCDAB89u));  // xor 1
    v = fmaxf(v, permrow16(v, 0x45670123u, 0xCDEF89ABu));  // xor 3
    v = fmaxf(v, permrow16(v, 0x01234567u, 0x89ABCDEFu));  // xor 7
    v = fmaxf(v, permrow16(v, 0xFEDCBA98u, 0x76543210u));  // xor 8
    return v;
}
__device__ __forceinline__ float rowred_sum(float v) {
    v += permrow16(v, 0x67452301u, 0xEFCDAB89u);
    v += permrow16(v, 0x45670123u, 0xCDEF89ABu);
    v += permrow16(v, 0x01234567u, 0x89ABCDEFu);
    v += permrow16(v, 0xFEDCBA98u, 0x76543210u);
    return v;
}

// Load a 16x32 f16 A/B fragment from row-major [*, 64] f16 storage.
// Documented CDNA5 16-bit A layout: lane<16 holds K in {kb..kb+7, kb+16..kb+23},
// lane>=16 the same +8.  Elements 0..7 <-> first b128, 8..15 <-> second b128.
__device__ __forceinline__ v16h ld_frag(const _Float16* row, int half16, int kbase) {
    const v8h lo = *reinterpret_cast<const v8h*>(row + kbase + 8 * half16);
    const v8h hi = *reinterpret_cast<const v8h*>(row + kbase + 8 * half16 + 16);
    v16h r;
#pragma unroll
    for (int i = 0; i < 8; ++i) { r[i] = lo[i]; r[i + 8] = hi[i]; }
    return r;
}

// ---------------------------------------------------------------------------
// 1x1-conv QKV projection.  x:[B,64,4096] f32 -> Q,K:[B,4096,64] f16 (row = d,
// contiguous, so WMMA fragments load as b128 pairs), V:[B,64,4096] f16
// (m contiguous, matching the P·V B-fragment access pattern).
// ---------------------------------------------------------------------------
__global__ void __launch_bounds__(256, 1)
proj_qkv(const float* __restrict__ xin,
         const float* __restrict__ qw, const float* __restrict__ qb,
         const float* __restrict__ kw, const float* __restrict__ kb,
         const float* __restrict__ vw, const float* __restrict__ vb,
         _Float16* __restrict__ Q, _Float16* __restrict__ K,
         _Float16* __restrict__ V) {
    const int idx = blockIdx.x * 256 + threadIdx.x;  // B*4096*64
    const int o = idx & 63;
    const int m = (idx >> 6) & 4095;
    const int b = idx >> 18;
    const float* xb = xin + ((size_t)b << 18);
    float aq = qb[o], ak = kb[o], av = vb[o];
#pragma unroll 4
    for (int i = 0; i < 64; ++i) {
        const float xv = xb[i * 4096 + m];   // broadcast across the o-group
        aq = fmaf(qw[o * 64 + i], xv, aq);
        ak = fmaf(kw[o * 64 + i], xv, ak);
        av = fmaf(vw[o * 64 + i], xv, av);
    }
    Q[((size_t)b * 4096 + m) * 64 + o] = (_Float16)aq;
    K[((size_t)b * 4096 + m) * 64 + o] = (_Float16)ak;
    V[((size_t)(b * 64 + o)) * 4096 + m] = (_Float16)av;
}

// ---------------------------------------------------------------------------
// Flash attention: one wave per 16-row Q tile, stream keys in chunks of 64.
// Online softmax; P staged through LDS to convert C-fragment -> A-fragment
// (same-wave LDS ops are in-order, no barrier required).
// ---------------------------------------------------------------------------
__global__ void __launch_bounds__(256, 1)
flash_attn(const _Float16* __restrict__ Q, const _Float16* __restrict__ K,
           const _Float16* __restrict__ V, float* __restrict__ Z) {
    __shared__ __align__(16) _Float16 pbuf[8 * 16 * 64];  // 16 KB: 8 waves x 16x64 P tile

    const int wave = threadIdx.x >> 5;
    const int lane = threadIdx.x & 31;
    const int half16 = lane >> 4;
    const int lrow = lane & 15;
    const int tile = blockIdx.x * 8 + wave;  // 0..1023 = B(4) * 256 tiles
    const int b = tile >> 8;
    const int tn = tile & 255;

    const _Float16* qr = Q + ((size_t)b * 4096 + tn * 16 + lrow) * 64;
    const v16h qa0 = ld_frag(qr, half16, 0);
    const v16h qa1 = ld_frag(qr, half16, 32);

    v8f mrun, lrun;
    v8f oacc[4];
#pragma unroll
    for (int r = 0; r < 8; ++r) {
        mrun[r] = -3.0e38f;
        lrun[r] = 0.f;
        oacc[0][r] = 0.f; oacc[1][r] = 0.f; oacc[2][r] = 0.f; oacc[3][r] = 0.f;
    }

    _Float16* pmine = pbuf + wave * (16 * 64);

#pragma unroll 1
    for (int mc = 0; mc < 64; ++mc) {   // 64 chunks of 64 keys = 4096
        const int m0 = mc * 64;

        // ---- S = Q * K^T for four 16-key tiles (no 1/sqrt(d) in reference)
        v8f s[4];
#pragma unroll
        for (int nt = 0; nt < 4; ++nt) {
            const _Float16* kr = K + ((size_t)b * 4096 + m0 + nt * 16 + lrow) * 64;
            const v16h kb0 = ld_frag(kr, half16, 0);
            const v16h kb1 = ld_frag(kr, half16, 32);
            v8f acc = {};
            acc = __builtin_amdgcn_wmma_f32_16x16x32_f16(false, qa0, false, kb0,
                                                         (short)0, acc, false, false);
            acc = __builtin_amdgcn_wmma_f32_16x16x32_f16(false, qa1, false, kb1,
                                                         (short)0, acc, false, false);
            s[nt] = acc;
        }

        // ---- online softmax.  C-fragment: element r <-> row r+8*half16,
        //      lane&15 <-> column, so row reductions stay in 16-lane rows
        //      and map onto v_permlane16_b32 butterflies (pure VALU).
        v8f mx = s[0];
#pragma unroll
        for (int nt = 1; nt < 4; ++nt)
#pragma unroll
            for (int r = 0; r < 8; ++r) mx[r] = fmaxf(mx[r], s[nt][r]);
#pragma unroll
        for (int r = 0; r < 8; ++r) mx[r] = rowred_max(mx[r]);

        v8f alpha;
#pragma unroll
        for (int r = 0; r < 8; ++r) {
            const float mn = fmaxf(mrun[r], mx[r]);
            alpha[r] = __expf(mrun[r] - mn);
            mrun[r] = mn;
        }
        v8f rs = {};
#pragma unroll
        for (int nt = 0; nt < 4; ++nt)
#pragma unroll
            for (int r = 0; r < 8; ++r) {
                const float p = __expf(s[nt][r] - mrun[r]);
                s[nt][r] = p;
                rs[r] += p;
            }
#pragma unroll
        for (int r = 0; r < 8; ++r) rs[r] = rowred_sum(rs[r]);
#pragma unroll
        for (int r = 0; r < 8; ++r) lrun[r] = lrun[r] * alpha[r] + rs[r];
#pragma unroll
        for (int ct = 0; ct < 4; ++ct)
#pragma unroll
            for (int r = 0; r < 8; ++r) oacc[ct][r] *= alpha[r];

        // ---- stage P (f16) to LDS, reload as A fragments (layout transpose)
#pragma unroll
        for (int nt = 0; nt < 4; ++nt)
#pragma unroll
            for (int r = 0; r < 8; ++r)
                pmine[(r + 8 * half16) * 64 + nt * 16 + lrow] = (_Float16)s[nt][r];
        const v16h pa0 = ld_frag(pmine + lrow * 64, half16, 0);
        const v16h pa1 = ld_frag(pmine + lrow * 64, half16, 32);

        // ---- O += P * V^T : contract over m, V[c, m] m-contiguous
#pragma unroll
        for (int ct = 0; ct < 4; ++ct) {
            const _Float16* vr = V + ((size_t)(b * 64 + ct * 16 + lrow)) * 4096 + m0;
            const v16h vb0 = ld_frag(vr, half16, 0);
            const v16h vb1 = ld_frag(vr, half16, 32);
            oacc[ct] = __builtin_amdgcn_wmma_f32_16x16x32_f16(false, pa0, false, vb0,
                                                              (short)0, oacc[ct], false, false);
            oacc[ct] = __builtin_amdgcn_wmma_f32_16x16x32_f16(false, pa1, false, vb1,
                                                              (short)0, oacc[ct], false, false);
        }
    }

    // ---- normalize and write z as flat [B][N=4096][C=64] f32
#pragma unroll
    for (int ct = 0; ct < 4; ++ct)
#pragma unroll
        for (int r = 0; r < 8; ++r) {
            const int n = tn * 16 + r + 8 * half16;
            Z[((size_t)b * 4096 + n) * 64 + ct * 16 + lrow] = oacc[ct][r] / lrun[r];
        }
}

// ---------------------------------------------------------------------------
// Attention epilogue: xa = conv1x1(reshape(z) * x, fw, fb) + x.
// The reference does a RAW reshape [B,HW,C]->[B,C,H,W]; reading the flat Z
// buffer as [B][64][4096] reproduces that scramble exactly.
// ---------------------------------------------------------------------------
__global__ void __launch_bounds__(256, 1)
att_epilogue(const float* __restrict__ Z, const float* __restrict__ xin,
             const float* __restrict__ fw, const float* __restrict__ fb,
             float* __restrict__ outa) {
    const int idx = blockIdx.x * 256 + threadIdx.x;  // B*64*4096
    const int m = idx & 4095;
    const int o = (idx >> 12) & 63;
    const int b = idx >> 18;
    const size_t base = (size_t)b << 18;
    float acc = fb[o];
#pragma unroll 4
    for (int i = 0; i < 64; ++i) {
        const float zz = Z[base + (size_t)i * 4096 + m];
        const float xv = xin[base + (size_t)i * 4096 + m];
        acc = fmaf(fw[o * 64 + i], zz * xv, acc);
    }
    outa[base + (size_t)o * 4096 + m] = acc + xin[base + (size_t)o * 4096 + m];
}

// ---------------------------------------------------------------------------
// Direct 3x3 conv, pad 1: in = concat(xa, ha) -> 128 ch, out 256 ch.
// Block = 16x16 pixel tile x 16 output channels; inputs staged 8 ch at a time
// through LDS (18x18x8 = 10.1 KB), weights via scalar (uniform) loads.
// ---------------------------------------------------------------------------
__global__ void __launch_bounds__(256, 1)
conv3x3(const float* __restrict__ xa, const float* __restrict__ ha,
        const float* __restrict__ w, const float* __restrict__ bias,
        float* __restrict__ y) {
    __shared__ float patch[8][18][18];
    const int t = threadIdx.x;
    const int tile = blockIdx.x;  // 0..15 -> 4x4 tiles of 16x16
    const int ocg = blockIdx.y;   // 0..15 -> 16 output channels each
    const int b = blockIdx.z;     // 0..3
    const int ty = (tile >> 2) * 16, tx = (tile & 3) * 16;
    const int ly = t >> 4, lx = t & 15;

    float acc[16];
#pragma unroll
    for (int j = 0; j < 16; ++j) acc[j] = bias[ocg * 16 + j];

    for (int icc = 0; icc < 16; ++icc) {  // 16 chunks of 8 input channels
        for (int s = t; s < 8 * 18 * 18; s += 256) {
            const int ic8 = s / 324;
            const int rr = s - ic8 * 324;
            const int yy = rr / 18, xx = rr - yy * 18;
            const int gy = ty + yy - 1, gx = tx + xx - 1;
            float v = 0.f;
            if (gy >= 0 && gy < 64 && gx >= 0 && gx < 64) {
                const int ic = icc * 8 + ic8;
                const float* src = (ic < 64)
                    ? (xa + ((size_t)(b * 64 + ic)) * 4096)
                    : (ha + ((size_t)(b * 64 + ic - 64)) * 4096);
                v = src[gy * 64 + gx];
            }
            patch[ic8][yy][xx] = v;
        }
        __syncthreads();
#pragma unroll
        for (int ic8 = 0; ic8 < 8; ++ic8) {
            float v[9];
#pragma unroll
            for (int dy = 0; dy < 3; ++dy)
#pragma unroll
                for (int dx = 0; dx < 3; ++dx)
                    v[dy * 3 + dx] = patch[ic8][ly + dy][lx + dx];
#pragma unroll
            for (int j = 0; j < 16; ++j) {
                const float* wp = w + (((size_t)(ocg * 16 + j)) * 128 + icc * 8 + ic8) * 9;
                float a = acc[j];
#pragma unroll
                for (int k = 0; k < 9; ++k) a = fmaf(wp[k], v[k], a);
                acc[j] = a;
            }
        }
        __syncthreads();
    }
#pragma unroll
    for (int j = 0; j < 16; ++j)
        y[((size_t)b * 256 + ocg * 16 + j) * 4096 + (ty + ly) * 64 + tx + lx] = acc[j];
}

// ---------------------------------------------------------------------------
// GroupNorm(groups==channels): per-(b, ch) mean / rsqrt(var+eps) over 4096.
// ---------------------------------------------------------------------------
__global__ void __launch_bounds__(256, 1)
gn_stats(const float* __restrict__ y, float* __restrict__ stats) {
    __shared__ float s1[8], s2[8];
    const int bc = blockIdx.x;  // b*256 + ch
    const float* p = y + (size_t)bc * 4096;
    float sum = 0.f, sq = 0.f;
    for (int i = threadIdx.x; i < 4096; i += 256) {
        const float v = p[i];
        sum += v;
        sq = fmaf(v, v, sq);
    }
#pragma unroll
    for (int d = 16; d >= 1; d >>= 1) {
        sum += __shfl_xor(sum, d, 32);
        sq += __shfl_xor(sq, d, 32);
    }
    if ((threadIdx.x & 31) == 0) { s1[threadIdx.x >> 5] = sum; s2[threadIdx.x >> 5] = sq; }
    __syncthreads();
    if (threadIdx.x == 0) {
        float S = 0.f, Qs = 0.f;
#pragma unroll
        for (int i = 0; i < 8; ++i) { S += s1[i]; Qs += s2[i]; }
        const float mean = S * (1.f / 4096.f);
        const float var = Qs * (1.f / 4096.f) - mean * mean;
        stats[bc * 2] = mean;
        stats[bc * 2 + 1] = rsqrtf(var + 1e-5f);
    }
}

// ---------------------------------------------------------------------------
// Gates: i,f,o,g = channel quarters of GN(y);
// c' = sigmoid(f)*c + sigmoid(i)*tanh(g); h' = sigmoid(o)*tanh(c').
// out = [h_next ; c_next], each B*64*4096 floats.
// ---------------------------------------------------------------------------
__global__ void __launch_bounds__(256, 1)
gates_kernel(const float* __restrict__ y, const float* __restrict__ stats,
             const float* __restrict__ gw, const float* __restrict__ gb,
             const float* __restrict__ cprev, float* __restrict__ out) {
    const int idx = blockIdx.x * 256 + threadIdx.x;  // B*64*4096
    const int b = idx >> 18;
    const int r = idx & 262143;
    const int ch = r >> 12;
    const int hw = r & 4095;
    float g4[4];
#pragma unroll
    for (int k = 0; k < 4; ++k) {
        const int cc = ch + 64 * k;
        const int bc = b * 256 + cc;
        float v = y[(size_t)bc * 4096 + hw];
        v = (v - stats[bc * 2]) * stats[bc * 2 + 1];
        g4[k] = v * gw[cc] + gb[cc];
    }
    const float ig = 1.f / (1.f + __expf(-g4[0]));
    const float fg = 1.f / (1.f + __expf(-g4[1]));
    const float og = 1.f / (1.f + __expf(-g4[2]));
    const float gg = tanhf(g4[3]);
    const float cn = fg * cprev[idx] + ig * gg;
    out[idx] = og * tanhf(cn);          // h_next
    out[idx + (1 << 20)] = cn;          // c_next (offset B*64*4096)
}

// ---------------------------------------------------------------------------
extern "C" void kernel_launch(void* const* d_in, const int* in_sizes, int n_in,
                              void* d_out, int out_size, void* d_ws, size_t ws_size,
                              hipStream_t stream) {
    const float* x = (const float*)d_in[0];
    const float* h = (const float*)d_in[1];
    const float* c = (const float*)d_in[2];

    char* ws = (char*)d_ws;
    size_t off = 0;
    auto carve = [&](size_t bytes) -> void* {
        void* p = ws + off;
        off += (bytes + 255) & ~(size_t)255;
        return p;
    };
    const size_t NE = 4ull * 4096 * 64;  // 1,048,576 elements per [B,64,4096] tensor

    _Float16* Qx = (_Float16*)carve(NE * 2);
    _Float16* Kx = (_Float16*)carve(NE * 2);
    _Float16* Vx = (_Float16*)carve(NE * 2);
    _Float16* Qh = (_Float16*)carve(NE * 2);
    _Float16* Kh = (_Float16*)carve(NE * 2);
    _Float16* Vh = (_Float16*)carve(NE * 2);
    float* Zx = (float*)carve(NE * 4);
    float* Zh = (float*)carve(NE * 4);
    float* xa = (float*)carve(NE * 4);
    float* ha = (float*)carve(NE * 4);
    float* y = (float*)carve(4ull * 256 * 4096 * 4);
    float* stats = (float*)carve(2048 * 4);

    proj_qkv<<<4096, 256, 0, stream>>>(
        x, (const float*)d_in[3], (const float*)d_in[4], (const float*)d_in[5],
        (const float*)d_in[6], (const float*)d_in[7], (const float*)d_in[8],
        Qx, Kx, Vx);
    proj_qkv<<<4096, 256, 0, stream>>>(
        h, (const float*)d_in[11], (const float*)d_in[12], (const float*)d_in[13],
        (const float*)d_in[14], (const float*)d_in[15], (const float*)d_in[16],
        Qh, Kh, Vh);

    flash_attn<<<128, 256, 0, stream>>>(Qx, Kx, Vx, Zx);
    flash_attn<<<128, 256, 0, stream>>>(Qh, Kh, Vh, Zh);

    att_epilogue<<<4096, 256, 0, stream>>>(Zx, x, (const float*)d_in[9],
                                           (const float*)d_in[10], xa);
    att_epilogue<<<4096, 256, 0, stream>>>(Zh, h, (const float*)d_in[17],
                                           (const float*)d_in[18], ha);

    conv3x3<<<dim3(16, 16, 4), 256, 0, stream>>>(
        xa, ha, (const float*)d_in[19], (const float*)d_in[20], y);

    gn_stats<<<1024, 256, 0, stream>>>(y, stats);
    gates_kernel<<<4096, 256, 0, stream>>>(y, stats, (const float*)d_in[21],
                                           (const float*)d_in[22], c, (float*)d_out);
}
